// ContrastiveLoss_65635690218207
// MI455X (gfx1250) — matrix operand order repaired and checked by reading, not compiled, not run on hardware
//
#include <hip/hip_runtime.h>
#include <cstdint>

// ---------------- types ----------------
typedef __attribute__((ext_vector_type(16))) __bf16 v16bf;
typedef __attribute__((ext_vector_type(8)))  __bf16 v8bf;
typedef __attribute__((ext_vector_type(4)))  __bf16 v4bf;
typedef __attribute__((ext_vector_type(8)))  float  v8f;

union BF16x16 {
    v16bf v;
    struct { v8bf lo; v8bf hi; } h;
};

#define NROWS   8192
#define DIM     1024
#define ROWPAD  1032                  // 1024 + 8 bf16 pad -> conflict-free ds_load_b128
#define LDS_BYTES (128 * ROWPAD * 2)  // 264192 bytes (<= 320KB WGP LDS)
#define TEMP_INV  14.28571428571428571f  // 1 / 0.07
#define SQRT_TINV 3.77964473009227227f   // sqrt(1/0.07): folded into zn so A*B is pre-scaled by 1/T

// ---------------- 1) normalize rows, emit bf16 zn (pre-scaled by sqrt(1/T)) + fp32 inv-norm ----
extern "C" __global__ __launch_bounds__(256)
void nt_normalize(const float* __restrict__ zi, const float* __restrict__ zj,
                  __bf16* __restrict__ zn, float* __restrict__ invn) {
    const int row = blockIdx.x;                       // 8192 rows
    const float* src = (row < NROWS / 2) ? (zi + row * DIM)
                                         : (zj + (row - NROWS / 2) * DIM);
    const int t = threadIdx.x;                        // 256 threads, 4 floats each
    float4 v = ((const float4*)src)[t];
    float ss = v.x * v.x + v.y * v.y + v.z * v.z + v.w * v.w;
    #pragma unroll
    for (int mask = 16; mask >= 1; mask >>= 1) ss += __shfl_xor(ss, mask, 32);

    __shared__ float red[8];
    __shared__ float sInv;
    if ((t & 31) == 0) red[t >> 5] = ss;
    __syncthreads();
    if (t == 0) {
        float tot = 0.f;
        #pragma unroll
        for (int i = 0; i < 8; ++i) tot += red[i];
        float inv = 1.f / fmaxf(sqrtf(tot), 1e-8f);
        sInv = inv;
        invn[row] = inv;
    }
    __syncthreads();
    float s = sInv * SQRT_TINV;
    v4bf o;
    o[0] = (__bf16)(v.x * s);
    o[1] = (__bf16)(v.y * s);
    o[2] = (__bf16)(v.z * s);
    o[3] = (__bf16)(v.w * s);
    ((v4bf*)(zn + row * DIM))[t] = o;
}

// ---------------- 2) positive-pair logits in fp32 ----------------
extern "C" __global__ __launch_bounds__(256)
void nt_pos(const float* __restrict__ zi, const float* __restrict__ zj,
            const float* __restrict__ invn, float* __restrict__ pos) {
    const int w = threadIdx.x >> 5, lane = threadIdx.x & 31;
    const int row = blockIdx.x * 8 + w;               // 1024 blocks * 8 waves
    const int p = (row + NROWS / 2) & (NROWS - 1);
    const float* a = (row < NROWS / 2) ? (zi + row * DIM) : (zj + (row - NROWS / 2) * DIM);
    const float* b = (p   < NROWS / 2) ? (zi + p   * DIM) : (zj + (p   - NROWS / 2) * DIM);
    float d = 0.f;
    for (int k = lane; k < DIM; k += 32) d += a[k] * b[k];
    #pragma unroll
    for (int mask = 16; mask >= 1; mask >>= 1) d += __shfl_xor(d, mask, 32);
    if (lane == 0) pos[row] = d * invn[row] * invn[p] * TEMP_INV;
}

// ---------------- 3) fused WMMA GEMM + online logsumexp ----------------
// grid.x = 512: 64 row-groups (128 rows) x 8 column-splits (1024 cols).
// Wave w: row-quarter (w&3) -> 32 rows, column-half (w>>2) -> 512 cols.
// Each wave computes 32x32 macro-tiles: 2 A-frags + 2 B-frags -> 4 WMMAs (2 loads/WMMA).
extern "C" __global__ __launch_bounds__(256)
void nt_main(const __bf16* __restrict__ zn,
             float* __restrict__ mPart, float* __restrict__ sPart) {
    extern __shared__ __bf16 As[];                    // 128 rows * ROWPAD
    const int w    = threadIdx.x >> 5;
    const int lane = threadIdx.x & 31;
    const int rg   = blockIdx.x >> 3;                 // row group (0..63)
    const int cs   = blockIdx.x & 7;                  // column split (0..7)

    // Stage all 128 A-rows block-cooperatively: thread t -> row t/2, half t&1 (512 elems).
    {
        const int r = threadIdx.x >> 1, half = threadIdx.x & 1;
        const uint4* gsrc = (const uint4*)(zn + (rg * 128 + r) * DIM + half * 512);
        uint4* ldst = (uint4*)(As + r * ROWPAD + half * 512);
        #pragma unroll 4
        for (int i = 0; i < 64; ++i) ldst[i] = gsrc[i];
    }
    __syncthreads();

    const int rq   = w & 3;                           // row quarter: 32 rows
    const int ch   = w >> 2;                          // column half: 512 cols
    const int m15  = lane & 15;
    const int hi16 = lane >> 4;                       // 0: lanes 0-15, 1: lanes 16-31
    const int rOff = hi16 * 8;                        // C rows 0-7 vs 8-15
    const int rBaseGlob = rg * 128 + rq * 32;

    // A frag: lane m15 = row; K chunks {0..7,16..23} (lo) / {8..15,24..31} (hi).
    const __bf16* aBase0 = As + (rq * 32 + m15) * ROWPAD + hi16 * 8;
    const __bf16* aBase1 = aBase0 + 16 * ROWPAD;
    const int cbBase  = cs * 64 + ch * 32;            // first 16-col subtile index
    const int diagCb0 = rBaseGlob >> 4;               // diag subtile for rows 0-15
    const int diagCb1 = diagCb0 + 1;                  // diag subtile for rows 16-31

    float mAcc[16], sAcc[16];                         // per-lane state: 2 row-subtiles x 8 rows
    #pragma unroll
    for (int r = 0; r < 16; ++r) { mAcc[r] = -INFINITY; sAcc[r] = 0.f; }

    for (int ct = 0; ct < 16; ++ct) {                 // 16 macro-tiles of 32 columns
        const int cb0 = cbBase + ct * 2;
        // B frag: lane m15 = column-row; K chunks {0..7,16..23} (lo) / {8..15,24..31} (hi).
        const __bf16* bBase0 = zn + (cb0 * 16 + m15) * DIM + hi16 * 8;
        const __bf16* bBase1 = bBase0 + 16 * DIM;

        v8f acc[2][2];
        #pragma unroll
        for (int i = 0; i < 2; ++i)
            #pragma unroll
            for (int j = 0; j < 2; ++j)
                acc[i][j] = (v8f){0.f, 0.f, 0.f, 0.f, 0.f, 0.f, 0.f, 0.f};

        #pragma unroll 2
        for (int k0 = 0; k0 < DIM; k0 += 32) {
            BF16x16 a0, a1, b0, b1;
            a0.h.lo = *(const v8bf*)(aBase0 + k0);
            a0.h.hi = *(const v8bf*)(aBase0 + k0 + 16);
            a1.h.lo = *(const v8bf*)(aBase1 + k0);
            a1.h.hi = *(const v8bf*)(aBase1 + k0 + 16);
            b0.h.lo = *(const v8bf*)(bBase0 + k0);
            b0.h.hi = *(const v8bf*)(bBase0 + k0 + 16);
            b1.h.lo = *(const v8bf*)(bBase1 + k0);
            b1.h.hi = *(const v8bf*)(bBase1 + k0 + 16);
            acc[0][0] = __builtin_amdgcn_wmma_f32_16x16x32_bf16(
                false, a0.v, false, b0.v, (short)0, acc[0][0], false, false);
            acc[0][1] = __builtin_amdgcn_wmma_f32_16x16x32_bf16(
                false, a0.v, false, b1.v, (short)0, acc[0][1], false, false);
            acc[1][0] = __builtin_amdgcn_wmma_f32_16x16x32_bf16(
                false, a1.v, false, b0.v, (short)0, acc[1][0], false, false);
            acc[1][1] = __builtin_amdgcn_wmma_f32_16x16x32_bf16(
                false, a1.v, false, b1.v, (short)0, acc[1][1], false, false);
        }

        // Online logsumexp update (products already scaled by 1/T via SQRT_TINV folding).
        #pragma unroll
        for (int i = 0; i < 2; ++i) {
            const int diagCbI = i ? diagCb1 : diagCb0;
            #pragma unroll
            for (int j = 0; j < 2; ++j) {
                const bool diag = ((cb0 + j) == diagCbI);
                #pragma unroll
                for (int r = 0; r < 8; ++r) {
                    float v = acc[i][j][r];
                    if (diag && (m15 == r + rOff)) v = -1.0e30f;  // mask self-similarity
                    const int idx = i * 8 + r;
                    float mn = fmaxf(mAcc[idx], v);
                    sAcc[idx] = sAcc[idx] * __expf(mAcc[idx] - mn) + __expf(v - mn);
                    mAcc[idx] = mn;
                }
            }
        }
    }

    // Butterfly-merge the 16 column-lanes within each row half (masks < 16 stay in-half).
    #pragma unroll
    for (int mask = 1; mask < 16; mask <<= 1) {
        #pragma unroll
        for (int idx = 0; idx < 16; ++idx) {
            float mo = __shfl_xor(mAcc[idx], mask, 32);
            float so = __shfl_xor(sAcc[idx], mask, 32);
            float mn = fmaxf(mAcc[idx], mo);
            sAcc[idx] = sAcc[idx] * __expf(mAcc[idx] - mn) + so * __expf(mo - mn);
            mAcc[idx] = mn;
        }
    }
    if (m15 == 0) {
        const int slot = cs * 2 + ch;                 // 16 column-slices per row
        #pragma unroll
        for (int i = 0; i < 2; ++i)
            #pragma unroll
            for (int r = 0; r < 8; ++r) {
                int row = rBaseGlob + i * 16 + rOff + r;
                mPart[row * 16 + slot] = mAcc[i * 8 + r];
                sPart[row * 16 + slot] = sAcc[i * 8 + r];
            }
    }
}

// ---------------- 4) merge column-slices, lse - pos, block partials ----------------
extern "C" __global__ __launch_bounds__(256)
void nt_merge(const float* __restrict__ mPart, const float* __restrict__ sPart,
              const float* __restrict__ pos, float* __restrict__ partial) {
    const int row = blockIdx.x * 256 + threadIdx.x;   // 32 blocks x 256 = 8192
    float m = mPart[row * 16], s = sPart[row * 16];
    #pragma unroll
    for (int c = 1; c < 16; ++c) {
        float mo = mPart[row * 16 + c], so = sPart[row * 16 + c];
        float mn = fmaxf(m, mo);
        s = s * __expf(m - mn) + so * __expf(mo - mn);
        m = mn;
    }
    float contrib = (m + __logf(s)) - pos[row];

    #pragma unroll
    for (int mask = 16; mask >= 1; mask >>= 1) contrib += __shfl_xor(contrib, mask, 32);
    __shared__ float red[8];
    if ((threadIdx.x & 31) == 0) red[threadIdx.x >> 5] = contrib;
    __syncthreads();
    if (threadIdx.x == 0) {
        float t = 0.f;
        #pragma unroll
        for (int i = 0; i < 8; ++i) t += red[i];
        partial[blockIdx.x] = t;
    }
}

// ---------------- 5) final scalar ----------------
extern "C" __global__ __launch_bounds__(32)
void nt_final(const float* __restrict__ partial, float* __restrict__ out) {
    float v = partial[threadIdx.x];                   // 32 partials
    #pragma unroll
    for (int mask = 16; mask >= 1; mask >>= 1) v += __shfl_xor(v, mask, 32);
    if (threadIdx.x == 0) out[0] = v * (1.0f / (float)NROWS);
}

// ---------------- launch ----------------
extern "C" void kernel_launch(void* const* d_in, const int* in_sizes, int n_in,
                              void* d_out, int out_size, void* d_ws, size_t ws_size,
                              hipStream_t stream) {
    const float* zi = (const float*)d_in[0];
    const float* zj = (const float*)d_in[1];

    char* ws = (char*)d_ws;
    __bf16* zn   = (__bf16*)ws;                                   // 16 MB
    float* invn  = (float*)(ws + (size_t)NROWS * DIM * 2);        // 32 KB
    float* pos   = invn + NROWS;                                  // 32 KB
    float* mPart = pos + NROWS;                                   // 512 KB
    float* sPart = mPart + NROWS * 16;                            // 512 KB
    float* partial = sPart + NROWS * 16;                          // 128 B
    float* out = (float*)d_out;

    (void)hipFuncSetAttribute((const void*)nt_main,
                              hipFuncAttributeMaxDynamicSharedMemorySize, LDS_BYTES);

    nt_normalize<<<NROWS, 256, 0, stream>>>(zi, zj, zn, invn);
    nt_pos<<<NROWS / 8, 256, 0, stream>>>(zi, zj, invn, pos);
    nt_main<<<512, 256, LDS_BYTES, stream>>>(zn, mPart, sPart);
    nt_merge<<<32, 256, 0, stream>>>(mPart, sPart, pos, partial);
    nt_final<<<1, 32, 0, stream>>>(partial, out);
}
